// Speaker_32435593019822
// MI455X (gfx1250) — matrix operand order repaired and checked by reading, not compiled, not run on hardware
//
#include <hip/hip_runtime.h>
#include <math.h>

#define B_SZ    16384
#define D_IMGSZ 2048
#define H_DIM   512
#define VOCAB   200
#define T_STEPS 20
#define VPAD    208   // vocab padded to multiple of 16
#define LPITCH  72    // LDS row pitch in bf16 elements for 64-wide K-chunks

typedef __bf16 bf16_t;
typedef __attribute__((ext_vector_type(16))) __bf16 v16bf;
typedef __attribute__((ext_vector_type(8)))  __bf16 v8bf;
typedef __attribute__((ext_vector_type(8)))  float   v8f;
typedef unsigned int v4u __attribute__((ext_vector_type(4)));
typedef int          v4i __attribute__((ext_vector_type(4)));
typedef int          v8i __attribute__((ext_vector_type(8)));

// ---------------------------------------------------------------------------
// WMMA fragment helpers (layouts per CDNA5 ISA 7.12.2, bf16 16x16x32)
//  A-frag: lane holds K = koff..koff+7 and koff+16..koff+23, koff = 8*(lane>>4)
//  B-frag: lane holds K = ks..ks+7 and ks+16..ks+23 (of W row n), ks = 8*(lane>>4)
// ---------------------------------------------------------------------------
__device__ __forceinline__ v16bf make_frag(const bf16_t* p0, const bf16_t* p1) {
  union { v16bf v; v8bf h[2]; } u;
  u.h[0] = *(const v8bf*)p0;
  u.h[1] = *(const v8bf*)p1;
  return u.v;
}
__device__ __forceinline__ v16bf load_a_frag(const bf16_t* row_ptr, int lane) {
  const int koff = (lane >> 4) * 8;
  return make_frag(row_ptr + koff, row_ptr + koff + 16);
}
__device__ __forceinline__ v16bf load_b_frag(const bf16_t* row_ptr, int lane) {
  const int ks = (lane >> 4) * 8;
  return make_frag(row_ptr + ks, row_ptr + ks + 16);
}
__device__ __forceinline__ v8f wmma_bf16(v16bf a, v16bf b, v8f c) {
  return __builtin_amdgcn_wmma_f32_16x16x32_bf16(false, a, false, b,
                                                 (short)0, c, false, false);
}
#define V8F_ZERO {0.f,0.f,0.f,0.f,0.f,0.f,0.f,0.f}

// ---------------------------------------------------------------------------
// Tensor Data Mover: DMA a rows x 64 bf16 tile (row stride ld_elems) from
// global into LDS at lds_byte_addr with a 72-element (144B) pitch, produced
// by the TDM pad feature: pad 4 DWORDs after every 32 DWORDs (one 128B row).
// D# packing per CDNA5 ISA 8.3/8.4.  Tracked by TENSORcnt.
// ---------------------------------------------------------------------------
__device__ __forceinline__ void tdm_load_tile64(unsigned lds_byte_addr,
                                                const bf16_t* gptr,
                                                int rows, int ld_elems)
{
  const unsigned long long ga = (unsigned long long)(size_t)gptr;
  v4u g0;
  g0[0] = 1u;                                        // count=1 (valid user D#)
  g0[1] = lds_byte_addr;                             // lds_addr
  g0[2] = (unsigned)(ga & 0xFFFFFFFFu);              // global_addr[31:0]
  g0[3] = (unsigned)((ga >> 32) & 0x01FFFFFFu)       // global_addr[56:32]
        | (2u << 30);                                // type = 2 ("image")
  const unsigned td0 = (unsigned)ld_elems;           // tensor_dim0 (elements)
  const unsigned td1 = (unsigned)rows;               // tensor_dim1
  const unsigned long long s0 = (unsigned long long)ld_elems; // dim0 stride
  v8i g1;
  g1[0] = (int)((1u << 16)      // data_size = 1 -> 2 bytes
              | (1u << 20)      // pad_enable
              | (4u << 22)      // pad_interval: 32 DWORDs (128B)
              | (3u << 25));    // pad_amount:   4 DWORDs (16B)
  g1[1] = (int)((td0 & 0xFFFFu) << 16);              // atomic_addr=0 | td0 lo
  g1[2] = (int)((td0 >> 16) | ((td1 & 0xFFFFu) << 16));
  g1[3] = (int)((td1 >> 16) | (64u << 16));          // tile_dim0 = 64
  g1[4] = (int)(rows & 0xFFFF);                      // tile_dim1 ; tile_dim2=0
  g1[5] = (int)(s0 & 0xFFFFFFFFu);                   // dim0 stride lo
  g1[6] = (int)((s0 >> 32) & 0xFFFFu);               // dim0 stride hi
  g1[7] = 0;
  const v4i gz = {0, 0, 0, 0};
#if defined(__clang_major__) && (__clang_major__ >= 23)
  const v8i gz8 = {0, 0, 0, 0, 0, 0, 0, 0};
  __builtin_amdgcn_tensor_load_to_lds(g0, g1, gz, gz, gz8, 0);
#else
  __builtin_amdgcn_tensor_load_to_lds(g0, g1, gz, gz, 0);
#endif
}

// ---------------------------------------------------------------------------
// Generic bf16 GEMM:  out[M,N] = epi( A[M,K] @ W[N,K]^T + bias[N] )
// EPI: 0 = identity, 1 = relu.  Block tile 128x128, 8 waves (4x2),
// wave tile 32x64.  K chunked by 64; tiles staged by TDM, double-buffered.
// M,N multiples of 128; K multiple of 64.
// ---------------------------------------------------------------------------
template <int EPI, typename OutT>
__global__ __launch_bounds__(256)
void gemm_bias_kernel(const bf16_t* __restrict__ A, const bf16_t* __restrict__ W,
                      const float* __restrict__ bias, OutT* __restrict__ out,
                      int M, int N, int K)
{
  __shared__ bf16_t lA[2][128 * LPITCH];
  __shared__ bf16_t lB[2][128 * LPITCH];

  const int tid   = threadIdx.x;
  const int lane  = tid & 31;
  const int wave  = tid >> 5;
  const int waveM = wave >> 1;
  const int waveN = wave & 1;
  const int m0    = blockIdx.x * 128;
  const int n0    = blockIdx.y * 128;
  const int frow  = lane & 15;
  (void)M;

  v8f acc[2][4];
  #pragma unroll
  for (int mi = 0; mi < 2; ++mi)
    #pragma unroll
    for (int ni = 0; ni < 4; ++ni)
      acc[mi][ni] = (v8f)V8F_ZERO;

  const unsigned la0 = (unsigned)(size_t)&lA[0][0];
  const unsigned la1 = (unsigned)(size_t)&lA[1][0];
  const unsigned lb0 = (unsigned)(size_t)&lB[0][0];
  const unsigned lb1 = (unsigned)(size_t)&lB[1][0];

  if (wave == 0) {
    tdm_load_tile64(la0, A + (size_t)m0 * K, 128, K);
    tdm_load_tile64(lb0, W + (size_t)n0 * K, 128, K);
  }
  int buf = 0;
  for (int k0 = 0; k0 < K; k0 += 64, buf ^= 1) {
    if (wave == 0) {
      if (k0 + 64 < K) {
        tdm_load_tile64(buf ? la0 : la1, A + (size_t)m0 * K + k0 + 64, 128, K);
        tdm_load_tile64(buf ? lb0 : lb1, W + (size_t)n0 * K + k0 + 64, 128, K);
        __builtin_amdgcn_s_wait_tensorcnt(2);
      } else {
        __builtin_amdgcn_s_wait_tensorcnt(0);
      }
    }
    __syncthreads();

    #pragma unroll
    for (int kk = 0; kk < 2; ++kk) {       // two 32-wide WMMA k-steps
      v16bf af[2], bfr[4];
      #pragma unroll
      for (int mi = 0; mi < 2; ++mi)
        af[mi] = load_a_frag(
            &lA[buf][(waveM * 32 + mi * 16 + frow) * LPITCH + kk * 32], lane);
      #pragma unroll
      for (int ni = 0; ni < 4; ++ni)
        bfr[ni] = load_b_frag(
            &lB[buf][(waveN * 64 + ni * 16 + frow) * LPITCH + kk * 32], lane);
      #pragma unroll
      for (int mi = 0; mi < 2; ++mi)
        #pragma unroll
        for (int ni = 0; ni < 4; ++ni)
          acc[mi][ni] = wmma_bf16(af[mi], bfr[ni], acc[mi][ni]);
    }

    __syncthreads();
  }

  const int colIn   = lane & 15;
  const int rowHalf = (lane >> 4) * 8;
  #pragma unroll
  for (int mi = 0; mi < 2; ++mi) {
    #pragma unroll
    for (int ni = 0; ni < 4; ++ni) {
      const int col = n0 + waveN * 64 + ni * 16 + colIn;
      const float bv = bias[col];
      #pragma unroll
      for (int j = 0; j < 8; ++j) {
        const int row = m0 + waveM * 32 + mi * 16 + rowHalf + j;
        float v = acc[mi][ni][j] + bv;
        if (EPI == 1) v = fmaxf(v, 0.f);
        out[(size_t)row * N + col] = (OutT)v;
      }
    }
  }
}

// ---------------------------------------------------------------------------
// Fused recurrent step:  hg = hx @ whh^T (three gate tiles r,z,n computed by
// the same block) with the full GRU gate math in the epilogue.
// Block tile: 128 rows x 64 h-cols; 8 waves (4x2), wave tile 32x32 per gate.
// hx ping-pongs (in != out) to avoid cross-block races.
// ---------------------------------------------------------------------------
__global__ __launch_bounds__(256)
void gru_step_kernel(const bf16_t* __restrict__ hxb_in,
                     const bf16_t* __restrict__ whh,
                     const float*  __restrict__ bhh,
                     const float*  __restrict__ xg,
                     const float*  __restrict__ hxf_in,
                     float*        __restrict__ hxf_out,
                     bf16_t*       __restrict__ hxb_out)
{
  __shared__ bf16_t lA[2][128 * LPITCH];
  __shared__ bf16_t lB[2][3][64 * LPITCH];

  const int tid   = threadIdx.x;
  const int lane  = tid & 31;
  const int wave  = tid >> 5;
  const int waveM = wave >> 1;           // 0..3 -> 32-row strip
  const int waveN = wave & 1;            // 0..1 -> 32-col strip
  const int m0    = blockIdx.x * 128;
  const int n0    = blockIdx.y * 64;
  const int frow  = lane & 15;

  v8f acc[3][2][2];                      // [gate r,z,n][mi][ni]
  #pragma unroll
  for (int g = 0; g < 3; ++g)
    #pragma unroll
    for (int mi = 0; mi < 2; ++mi)
      #pragma unroll
      for (int ni = 0; ni < 2; ++ni)
        acc[g][mi][ni] = (v8f)V8F_ZERO;

  unsigned la[2], lb[2][3];
  la[0] = (unsigned)(size_t)&lA[0][0];
  la[1] = (unsigned)(size_t)&lA[1][0];
  #pragma unroll
  for (int b = 0; b < 2; ++b)
    #pragma unroll
    for (int g = 0; g < 3; ++g)
      lb[b][g] = (unsigned)(size_t)&lB[b][g][0];

  if (wave == 0) {
    tdm_load_tile64(la[0], hxb_in + (size_t)m0 * H_DIM, 128, H_DIM);
    #pragma unroll
    for (int g = 0; g < 3; ++g)
      tdm_load_tile64(lb[0][g], whh + (size_t)(g * H_DIM + n0) * H_DIM, 64, H_DIM);
  }
  int buf = 0;
  for (int k0 = 0; k0 < H_DIM; k0 += 64, buf ^= 1) {
    if (wave == 0) {
      if (k0 + 64 < H_DIM) {
        tdm_load_tile64(la[buf ^ 1], hxb_in + (size_t)m0 * H_DIM + k0 + 64, 128, H_DIM);
        #pragma unroll
        for (int g = 0; g < 3; ++g)
          tdm_load_tile64(lb[buf ^ 1][g],
                          whh + (size_t)(g * H_DIM + n0) * H_DIM + k0 + 64, 64, H_DIM);
        __builtin_amdgcn_s_wait_tensorcnt(4);
      } else {
        __builtin_amdgcn_s_wait_tensorcnt(0);
      }
    }
    __syncthreads();

    #pragma unroll
    for (int kk = 0; kk < 2; ++kk) {
      v16bf af[2];
      #pragma unroll
      for (int mi = 0; mi < 2; ++mi)
        af[mi] = load_a_frag(
            &lA[buf][(waveM * 32 + mi * 16 + frow) * LPITCH + kk * 32], lane);
      #pragma unroll
      for (int g = 0; g < 3; ++g)
        #pragma unroll
        for (int ni = 0; ni < 2; ++ni) {
          const v16bf bfr = load_b_frag(
              &lB[buf][g][(waveN * 32 + ni * 16 + frow) * LPITCH + kk * 32], lane);
          #pragma unroll
          for (int mi = 0; mi < 2; ++mi)
            acc[g][mi][ni] = wmma_bf16(af[mi], bfr, acc[g][mi][ni]);
        }
    }

    __syncthreads();
  }

  // GRU gate epilogue
  const int colIn   = lane & 15;
  const int rowHalf = (lane >> 4) * 8;
  #pragma unroll
  for (int mi = 0; mi < 2; ++mi) {
    #pragma unroll
    for (int ni = 0; ni < 2; ++ni) {
      const int col = n0 + waveN * 32 + ni * 16 + colIn;
      const float br = bhh[col];
      const float bz = bhh[H_DIM + col];
      const float bn = bhh[2 * H_DIM + col];
      #pragma unroll
      for (int j = 0; j < 8; ++j) {
        const int row = m0 + waveM * 32 + mi * 16 + rowHalf + j;
        const size_t xb = (size_t)row * (3 * H_DIM);
        const float rp = acc[0][mi][ni][j] + br + xg[xb + col];
        const float zp = acc[1][mi][ni][j] + bz + xg[xb + H_DIM + col];
        const float hn = acc[2][mi][ni][j] + bn;
        const float r  = 1.f / (1.f + expf(-rp));
        const float z  = 1.f / (1.f + expf(-zp));
        const float n  = tanhf(xg[xb + 2 * H_DIM + col] + r * hn);
        const float hprev = hxf_in[(size_t)row * H_DIM + col];
        const float hnew  = (1.f - z) * n + z * hprev;
        hxf_out[(size_t)row * H_DIM + col] = hnew;
        hxb_out[(size_t)row * H_DIM + col] = (bf16_t)hnew;
      }
    }
  }
}

// ---------------------------------------------------------------------------
// Fused actor+critic heads for 32 batch rows per block (8 waves).
// ---------------------------------------------------------------------------
__global__ __launch_bounds__(256)
void heads_kernel(const bf16_t* __restrict__ hx_bf,
                  const bf16_t* __restrict__ wa1, const float* __restrict__ ba1,
                  const bf16_t* __restrict__ wa2, const float* __restrict__ ba2,
                  const bf16_t* __restrict__ wa3, const float* __restrict__ ba3,
                  const bf16_t* __restrict__ wc1, const float* __restrict__ bc1,
                  const bf16_t* __restrict__ wc2, const float* __restrict__ bc2,
                  const float*  __restrict__ wc3, const float* __restrict__ bc3,
                  const int*    __restrict__ actions,
                  float* __restrict__ out_lp, float* __restrict__ out_ent,
                  float* __restrict__ out_v, int t)
{
  __shared__ bf16_t l1[2][32 * 72];
  __shared__ bf16_t l2[2][32 * 72];
  __shared__ float  lg[32 * 216];

  const int tid   = threadIdx.x;
  const int lane  = tid & 31;
  const int wave  = tid >> 5;
  const int half  = wave >> 2;           // 0 = actor, 1 = critic
  const int cg    = wave & 3;            // 16-column group within 64
  const int b0    = blockIdx.x * 32;
  const int frow  = lane & 15;
  const int colIn = frow;
  const int rowHalf = (lane >> 4) * 8;

  const bf16_t* W1 = half ? wc1 : wa1;
  const bf16_t* W2 = half ? wc2 : wa2;
  const float*  B1 = half ? bc1 : ba1;
  const float*  B2 = half ? bc2 : ba2;

  // ---- stage 1: [32x512] @ [64x512]^T ----
  v8f s1[2] = { (v8f)V8F_ZERO, (v8f)V8F_ZERO };
  for (int k0 = 0; k0 < H_DIM; k0 += 32) {
    const v16bf bfr = load_b_frag(W1 + (size_t)(cg * 16 + frow) * H_DIM + k0, lane);
    #pragma unroll
    for (int mi = 0; mi < 2; ++mi) {
      const v16bf afr =
          load_a_frag(hx_bf + (size_t)(b0 + mi * 16 + frow) * H_DIM + k0, lane);
      s1[mi] = wmma_bf16(afr, bfr, s1[mi]);
    }
  }
  {
    const int col = cg * 16 + colIn;
    const float bv = B1[col];
    #pragma unroll
    for (int mi = 0; mi < 2; ++mi)
      #pragma unroll
      for (int j = 0; j < 8; ++j) {
        const int row = mi * 16 + rowHalf + j;
        l1[half][row * 72 + col] = (bf16_t)tanhf(s1[mi][j] + bv);
      }
  }
  __syncthreads();

  // ---- stage 2: [32x64] @ [64x64]^T ----
  v8f s2[2] = { (v8f)V8F_ZERO, (v8f)V8F_ZERO };
  #pragma unroll
  for (int k0 = 0; k0 < 64; k0 += 32) {
    const v16bf bfr = load_b_frag(W2 + (size_t)(cg * 16 + frow) * 64 + k0, lane);
    #pragma unroll
    for (int mi = 0; mi < 2; ++mi) {
      const v16bf afr = load_a_frag(&l1[half][(mi * 16 + frow) * 72 + k0], lane);
      s2[mi] = wmma_bf16(afr, bfr, s2[mi]);
    }
  }
  {
    const int col = cg * 16 + colIn;
    const float bv = B2[col];
    #pragma unroll
    for (int mi = 0; mi < 2; ++mi)
      #pragma unroll
      for (int j = 0; j < 8; ++j) {
        const int row = mi * 16 + rowHalf + j;
        l2[half][row * 72 + col] = (bf16_t)tanhf(s2[mi][j] + bv);
      }
  }
  __syncthreads();

  // ---- stage 3: logits [32x64] @ [208x64]^T ----
  for (int nt = wave; nt < VPAD / 16; nt += 8) {
    v8f sl[2] = { (v8f)V8F_ZERO, (v8f)V8F_ZERO };
    #pragma unroll
    for (int k0 = 0; k0 < 64; k0 += 32) {
      const v16bf bfr = load_b_frag(wa3 + (size_t)(nt * 16 + frow) * 64 + k0, lane);
      #pragma unroll
      for (int mi = 0; mi < 2; ++mi) {
        const v16bf afr = load_a_frag(&l2[0][(mi * 16 + frow) * 72 + k0], lane);
        sl[mi] = wmma_bf16(afr, bfr, sl[mi]);
      }
    }
    const int col = nt * 16 + colIn;
    const float bv = (col < VOCAB) ? ba3[col] : 0.f;
    #pragma unroll
    for (int mi = 0; mi < 2; ++mi)
      #pragma unroll
      for (int j = 0; j < 8; ++j) {
        const int row = mi * 16 + rowHalf + j;
        lg[row * 216 + col] = sl[mi][j] + bv;
      }
  }
  __syncthreads();

  // ---- value + log-softmax stats (one thread per row) ----
  if (tid < 32) {
    const int row = tid;
    const size_t gidx = (size_t)(b0 + row) * T_STEPS + t;

    float v = bc3[0];
    for (int k = 0; k < 64; ++k)
      v += (float)l2[1][row * 72 + k] * wc3[k];
    out_v[gidx] = v;

    float m = -1e30f;
    for (int j = 0; j < VOCAB; ++j) m = fmaxf(m, lg[row * 216 + j]);
    float Z = 0.f, S = 0.f;
    for (int j = 0; j < VOCAB; ++j) {
      const float x = lg[row * 216 + j] - m;
      const float e = expf(x);
      Z += e; S += e * x;
    }
    const float logZ = logf(Z);
    const int a = actions[gidx];
    out_lp[gidx]  = (lg[row * 216 + a] - m) - logZ;
    out_ent[gidx] = logZ - S / Z;
  }
}

// ---------------------------------------------------------------------------
// Small utility kernels
// ---------------------------------------------------------------------------
__global__ void cvt_bf16_kernel(bf16_t* __restrict__ dst,
                                const float* __restrict__ src, int n) {
  const int i = blockIdx.x * blockDim.x + threadIdx.x;
  if (i < n) dst[i] = (bf16_t)src[i];
}
__global__ void cvt_pad_a3_kernel(bf16_t* __restrict__ dst,
                                  const float* __restrict__ src) {
  const int i = blockIdx.x * blockDim.x + threadIdx.x;   // VPAD*64
  if (i >= VPAD * 64) return;
  const int n = i >> 6;
  dst[i] = (n < VOCAB) ? (bf16_t)src[i] : (bf16_t)0.f;
}
__global__ void zero_hx_kernel(float* __restrict__ hf,
                               bf16_t* __restrict__ hb, int n) {
  const int i = blockIdx.x * blockDim.x + threadIdx.x;
  if (i < n) { hf[i] = 0.f; hb[i] = (bf16_t)0.f; }
}
__global__ void actions_to_float_kernel(const int* __restrict__ a,
                                        float* __restrict__ o, int n) {
  const int i = blockIdx.x * blockDim.x + threadIdx.x;
  if (i < n) o[i] = (float)a[i];
}

// ---------------------------------------------------------------------------
extern "C" void kernel_launch(void* const* d_in, const int* in_sizes, int n_in,
                              void* d_out, int out_size, void* d_ws, size_t ws_size,
                              hipStream_t stream)
{
  (void)in_sizes; (void)n_in; (void)out_size; (void)ws_size;

  const float* images  = (const float*)d_in[0];
  const int*   actions = (const int*)  d_in[1];
  const float* enc_w1  = (const float*)d_in[2];
  const float* enc_b1  = (const float*)d_in[3];
  const float* enc_w2  = (const float*)d_in[4];
  const float* enc_b2  = (const float*)d_in[5];
  const float* gru_wih = (const float*)d_in[6];
  const float* gru_bih = (const float*)d_in[7];
  const float* gru_whh = (const float*)d_in[8];
  const float* gru_bhh = (const float*)d_in[9];
  const float* act_w1  = (const float*)d_in[10];
  const float* act_b1  = (const float*)d_in[11];
  const float* act_w2  = (const float*)d_in[12];
  const float* act_b2  = (const float*)d_in[13];
  const float* act_w3  = (const float*)d_in[14];
  const float* act_b3  = (const float*)d_in[15];
  const float* cri_w1  = (const float*)d_in[16];
  const float* cri_b1  = (const float*)d_in[17];
  const float* cri_w2  = (const float*)d_in[18];
  const float* cri_b2  = (const float*)d_in[19];
  const float* cri_w3  = (const float*)d_in[20];
  const float* cri_b3  = (const float*)d_in[21];

  char* ws = (char*)d_ws;
  size_t off = 0;
  auto alloc = [&](size_t bytes) -> char* {
    char* p = ws + off;
    off += (bytes + 255) & ~(size_t)255;
    return p;
  };

  bf16_t* img_bf = (bf16_t*)alloc((size_t)B_SZ * D_IMGSZ * 2);
  bf16_t* w_e1   = (bf16_t*)alloc((size_t)H_DIM * D_IMGSZ * 2);
  bf16_t* w_e2   = (bf16_t*)alloc((size_t)H_DIM * H_DIM * 2);
  bf16_t* w_ih   = (bf16_t*)alloc((size_t)3 * H_DIM * H_DIM * 2);
  bf16_t* w_hh   = (bf16_t*)alloc((size_t)3 * H_DIM * H_DIM * 2);
  bf16_t* w_a1   = (bf16_t*)alloc((size_t)64 * H_DIM * 2);
  bf16_t* w_a2   = (bf16_t*)alloc((size_t)64 * 64 * 2);
  bf16_t* w_a3   = (bf16_t*)alloc((size_t)VPAD * 64 * 2);
  bf16_t* w_c1   = (bf16_t*)alloc((size_t)64 * H_DIM * 2);
  bf16_t* w_c2   = (bf16_t*)alloc((size_t)64 * 64 * 2);
  bf16_t* x1_bf  = (bf16_t*)alloc((size_t)B_SZ * H_DIM * 2);
  bf16_t* x_bf   = (bf16_t*)alloc((size_t)B_SZ * H_DIM * 2);
  float*  xg     = (float*) alloc((size_t)B_SZ * 3 * H_DIM * 4);
  float*  hxf[2];
  bf16_t* hxb[2];
  hxf[0] = (float*) alloc((size_t)B_SZ * H_DIM * 4);
  hxf[1] = (float*) alloc((size_t)B_SZ * H_DIM * 4);
  hxb[0] = (bf16_t*)alloc((size_t)B_SZ * H_DIM * 2);
  hxb[1] = (bf16_t*)alloc((size_t)B_SZ * H_DIM * 2);

  float* out = (float*)d_out;
  const size_t BT = (size_t)B_SZ * T_STEPS;
  float* out_lp  = out + BT;
  float* out_ent = out + 2 * BT;
  float* out_v   = out + 3 * BT;

  const int THR = 256;
  auto blk = [](size_t n) { return (unsigned)((n + 255) / 256); };

  // --- bf16 conversions ---
  cvt_bf16_kernel<<<blk((size_t)B_SZ * D_IMGSZ), THR, 0, stream>>>(img_bf, images, B_SZ * D_IMGSZ);
  cvt_bf16_kernel<<<blk((size_t)H_DIM * D_IMGSZ), THR, 0, stream>>>(w_e1, enc_w1, H_DIM * D_IMGSZ);
  cvt_bf16_kernel<<<blk((size_t)H_DIM * H_DIM), THR, 0, stream>>>(w_e2, enc_w2, H_DIM * H_DIM);
  cvt_bf16_kernel<<<blk((size_t)3 * H_DIM * H_DIM), THR, 0, stream>>>(w_ih, gru_wih, 3 * H_DIM * H_DIM);
  cvt_bf16_kernel<<<blk((size_t)3 * H_DIM * H_DIM), THR, 0, stream>>>(w_hh, gru_whh, 3 * H_DIM * H_DIM);
  cvt_bf16_kernel<<<blk(64 * H_DIM), THR, 0, stream>>>(w_a1, act_w1, 64 * H_DIM);
  cvt_bf16_kernel<<<blk(64 * 64), THR, 0, stream>>>(w_a2, act_w2, 64 * 64);
  cvt_pad_a3_kernel<<<blk(VPAD * 64), THR, 0, stream>>>(w_a3, act_w3);
  cvt_bf16_kernel<<<blk(64 * H_DIM), THR, 0, stream>>>(w_c1, cri_w1, 64 * H_DIM);
  cvt_bf16_kernel<<<blk(64 * 64), THR, 0, stream>>>(w_c2, cri_w2, 64 * 64);

  // --- actions passthrough + h0 = 0 ---
  actions_to_float_kernel<<<blk(BT), THR, 0, stream>>>(actions, out, (int)BT);
  zero_hx_kernel<<<blk((size_t)B_SZ * H_DIM), THR, 0, stream>>>(hxf[0], hxb[0], B_SZ * H_DIM);

  // --- encoder + hoisted input gates ---
  gemm_bias_kernel<1, bf16_t><<<dim3(B_SZ / 128, H_DIM / 128), THR, 0, stream>>>(
      img_bf, w_e1, enc_b1, x1_bf, B_SZ, H_DIM, D_IMGSZ);
  gemm_bias_kernel<0, bf16_t><<<dim3(B_SZ / 128, H_DIM / 128), THR, 0, stream>>>(
      x1_bf, w_e2, enc_b2, x_bf, B_SZ, H_DIM, H_DIM);
  gemm_bias_kernel<0, float><<<dim3(B_SZ / 128, (3 * H_DIM) / 128), THR, 0, stream>>>(
      x_bf, w_ih, gru_bih, xg, B_SZ, 3 * H_DIM, H_DIM);

  // --- T recurrent steps (hx ping-pongs) ---
  for (int t = 0; t < T_STEPS; ++t) {
    const int cur = t & 1, nxt = cur ^ 1;
    gru_step_kernel<<<dim3(B_SZ / 128, H_DIM / 64), THR, 0, stream>>>(
        hxb[cur], w_hh, gru_bhh, xg, hxf[cur], hxf[nxt], hxb[nxt]);
    heads_kernel<<<B_SZ / 32, THR, 0, stream>>>(
        hxb[nxt], w_a1, act_b1, w_a2, act_b2, w_a3, act_b3,
        w_c1, cri_b1, w_c2, cri_b2, cri_w3, cri_b3,
        actions, out_lp, out_ent, out_v, t);
  }
}
